// LSTM_850403525108
// MI455X (gfx1250) — matrix operand order repaired and checked by reading, not compiled, not run on hardware
//
#include <hip/hip_runtime.h>
#include <math.h>

// ---- problem constants -----------------------------------------------------
#define BATCH   1024
#define SEQ     365
#define INSZ    64
#define HID     256
#define NG      1024      // 4*HID gate columns
#define ROWS    16        // batch rows per workgroup (one WMMA M tile)
#define NWG     (BATCH / ROWS)   // 64 workgroups

// workspace layout (bytes)
#define WS_BH_OFF   0                        // packed W_hh bf16: 1024*256*2 = 524288
#define WS_BX_OFF   524288                   // packed W_ih bf16: 1024*64*2  = 131072
#define WS_BIAS_OFF 655360                   // combined bias f32: 1024*4     = 4096

typedef __attribute__((ext_vector_type(16))) __bf16 v16bf;
typedef __attribute__((ext_vector_type(8)))  float  v8f;

union FragAB { int4 q[2]; v16bf v; };

__device__ __forceinline__ unsigned short f2bf(float f) {
    unsigned int u = __float_as_uint(f);
    unsigned int r = u + 0x7FFFu + ((u >> 16) & 1u);   // round-to-nearest-even
    return (unsigned short)(r >> 16);
}

// Hardware tanh if the builtin exists (CDNA5 lists TANH as a TRANS op);
// otherwise a safe exp-based fallback.
#if __has_builtin(__builtin_amdgcn_tanhf)
__device__ __forceinline__ float ftanh(float x) {
    return __builtin_amdgcn_tanhf(x);
}
#elif __has_builtin(__builtin_amdgcn_tanh_f32)
__device__ __forceinline__ float ftanh(float x) {
    return __builtin_amdgcn_tanh_f32(x);
}
#else
__device__ __forceinline__ float ftanh(float x) {
    float e = __expf(2.0f * x);
    return 1.0f - 2.0f / (e + 1.0f);    // safe at +-inf
}
#endif
// sigmoid(x) = 0.5 + 0.5*tanh(x/2): 3 VALU ops when tanh is a single TRANS op
__device__ __forceinline__ float fsigmoid(float x) {
    return fmaf(0.5f, ftanh(0.5f * x), 0.5f);
}

// ---- one-time weight pack + bias fuse --------------------------------------
// B-fragment layout (bf16 32x16 KxN): lane l holds column N = l&15;
// lanes 0-15 hold K = kc*32 + 0..15, lanes 16-31 hold K = kc*32 + 16..31,
// 16 bf16 (32 bytes) contiguous per lane -> fragment = 1024 bytes.
__global__ __launch_bounds__(256) void lstm_pack_kernel(
    const float* __restrict__ W_ih, const float* __restrict__ W_hh,
    const float* __restrict__ b_ih, const float* __restrict__ b_hh,
    unsigned short* __restrict__ Bh, unsigned short* __restrict__ Bx,
    float* __restrict__ bias)
{
    int idx = blockIdx.x * blockDim.x + threadIdx.x;

    // W_hh pack: 1024*256 = 262144 bf16 elements
    if (idx < NG * HID) {
        int p    = idx & 15;          // position within lane's 16 values
        int lane = (idx >> 4) & 31;
        int blk  = idx >> 9;          // fragment id = nt*8 + kc
        int kc   = blk & 7;
        int nt   = blk >> 3;
        int n = nt * 16 + (lane & 15);
        int k = kc * 32 + ((lane >> 4) & 1) * 16 + p;
        Bh[idx] = f2bf(W_hh[n * HID + k]);
    }
    // W_ih pack: 1024*64 = 65536 bf16 elements
    if (idx < NG * INSZ) {
        int p    = idx & 15;
        int lane = (idx >> 4) & 31;
        int blk  = idx >> 9;          // fragment id = nt*2 + kc
        int kc   = blk & 1;
        int nt   = blk >> 1;
        int n = nt * 16 + (lane & 15);
        int k = kc * 32 + ((lane >> 4) & 1) * 16 + p;
        Bx[idx] = f2bf(W_ih[n * INSZ + k]);
    }
    if (idx < NG) bias[idx] = b_ih[idx] + b_hh[idx];
}

// ---- persistent fused LSTM scan --------------------------------------------
__global__ __launch_bounds__(256) void lstm_scan_kernel(
    const float* __restrict__ inp,          // [B][T][IN]
    const unsigned short* __restrict__ Bh,  // packed W_hh bf16
    const unsigned short* __restrict__ Bx,  // packed W_ih bf16
    const float* __restrict__ bias,         // b_ih + b_hh, [4H]
    const float* __restrict__ W_lin,        // [1][HID]
    const float* __restrict__ b_lin,        // [1]
    float* __restrict__ out)                // [B][1]
{
    __shared__ unsigned short h_lds[ROWS][HID];   // bf16 hidden state, 8 KB
    __shared__ unsigned short x_lds[ROWS][INSZ];  // bf16 staged x_t,   2 KB
    __shared__ float          out_acc[ROWS];

    const int tid  = threadIdx.x;
    const int wave = tid >> 5;
    const int lane = tid & 31;
    const int b0   = blockIdx.x * ROWS;

    const int m     = lane & 15;          // A/D row within tile (lanes 0-15)
    const int half  = (lane >> 4) & 1;    // lane group
    const int nbase = wave * 32;          // hidden cols owned by this wave

    // zero hidden state + output accumulator
    for (int i = tid; i < ROWS * HID; i += 256) ((unsigned short*)h_lds)[i] = 0;
    if (tid < ROWS) out_acc[tid] = 0.0f;

    // per-lane constants: bias per (gate, tile), final linear weight per tile
    float biasv[4][2];
#pragma unroll
    for (int q = 0; q < 4; ++q)
#pragma unroll
        for (int tn = 0; tn < 2; ++tn)
            biasv[q][tn] = bias[q * HID + nbase + tn * 16 + (lane & 15)];
    float wlin[2];
#pragma unroll
    for (int tn = 0; tn < 2; ++tn)
        wlin[tn] = W_lin[nbase + tn * 16 + (lane & 15)];

    v8f cst[2];         // persistent cell state (f32, in VGPRs)
    cst[0] = {}; cst[1] = {};
    v8f hreg[2];        // last hidden values (f32)
    hreg[0] = {}; hreg[1] = {};

    __syncthreads();

#pragma unroll 1
    for (int t = 0; t < SEQ; ++t) {
        // Opaque copy of t: the compiler cannot prove the weight addresses are
        // loop-invariant, so B-fragment loads stay inside the loop as plain
        // global loads (L2/WGP$-resident) instead of being hoisted + spilled
        // to scratch (round-1 pathology).
        int t_op = t;
        asm volatile("" : "+s"(t_op));
        const size_t jitter = (size_t)(t_op >> 9);   // always 0, but opaque
        const int4* BhT = (const int4*)Bh + jitter;  // fragment = 64 int4 (1KB)
        const int4* BxT = (const int4*)Bx + jitter;

        // ---- stage x_t (fp32 -> bf16) into LDS: 16 rows x 64 cols ---------
        {
            int r  = tid >> 4;
            int c4 = (tid & 15) * 4;
            const float* src = inp + (size_t)(b0 + r) * (SEQ * INSZ)
                                   + (size_t)t * INSZ + c4;
            float4 xv = *(const float4*)src;
            unsigned int lo = (unsigned int)f2bf(xv.x) | ((unsigned int)f2bf(xv.y) << 16);
            unsigned int hi = (unsigned int)f2bf(xv.z) | ((unsigned int)f2bf(xv.w) << 16);
            uint2 pk; pk.x = lo; pk.y = hi;
            *(uint2*)&x_lds[r][c4] = pk;
        }
        __syncthreads();

        // ---- gates = bias + h @ W_hh^T + x @ W_ih^T  (WMMA bf16, f32 acc) -
        v8f acc[4][2];
#pragma unroll
        for (int q = 0; q < 4; ++q)
#pragma unroll
            for (int tn = 0; tn < 2; ++tn)
#pragma unroll
                for (int r = 0; r < 8; ++r)
                    acc[q][tn][r] = biasv[q][tn];

        // h contribution: K = 256 -> 8 chunks of 32.
        // Stage all 8 B fragments (16 x global_load_b128) ahead of the WMMAs;
        // the scheduler pipelines loads of chunk kc against WMMAs of kc-1.
#pragma unroll 1
        for (int kc = 0; kc < 8; ++kc) {
            FragAB a;
            int kb = kc * 32 + half * 16;
            a.q[0] = *(const int4*)&h_lds[m][kb];
            a.q[1] = *(const int4*)&h_lds[m][kb + 8];

            FragAB bm[8];
#pragma unroll
            for (int u = 0; u < 8; ++u) {
                int q  = u >> 1;
                int tn = u & 1;
                int nt = q * 16 + wave * 2 + tn;
                const int4* bp = BhT + (size_t)(nt * 8 + kc) * 64 + lane * 2;
                bm[u].q[0] = bp[0];
                bm[u].q[1] = bp[1];
            }
#pragma unroll
            for (int u = 0; u < 8; ++u) {
                int q  = u >> 1;
                int tn = u & 1;
                acc[q][tn] = __builtin_amdgcn_wmma_f32_16x16x32_bf16(
                    false, a.v, false, bm[u].v, (short)0, acc[q][tn], false, false);
            }
        }

        // x contribution: K = 64 -> 2 chunks of 32
#pragma unroll 1
        for (int kc = 0; kc < 2; ++kc) {
            FragAB a;
            int kb = kc * 32 + half * 16;
            a.q[0] = *(const int4*)&x_lds[m][kb];
            a.q[1] = *(const int4*)&x_lds[m][kb + 8];

            FragAB bm[8];
#pragma unroll
            for (int u = 0; u < 8; ++u) {
                int q  = u >> 1;
                int tn = u & 1;
                int nt = q * 16 + wave * 2 + tn;
                const int4* bp = BxT + (size_t)(nt * 2 + kc) * 64 + lane * 2;
                bm[u].q[0] = bp[0];
                bm[u].q[1] = bp[1];
            }
#pragma unroll
            for (int u = 0; u < 8; ++u) {
                int q  = u >> 1;
                int tn = u & 1;
                acc[q][tn] = __builtin_amdgcn_wmma_f32_16x16x32_bf16(
                    false, a.v, false, bm[u].v, (short)0, acc[q][tn], false, false);
            }
        }
        __syncthreads();   // all reads of h_lds / x_lds complete

        // ---- fused nonlinear update; i/f/g/o are all local to this wave ---
#pragma unroll
        for (int tn = 0; tn < 2; ++tn) {
            int j = nbase + tn * 16 + (lane & 15);
#pragma unroll
            for (int r = 0; r < 8; ++r) {
                float iv = fsigmoid(acc[0][tn][r]);
                float fv = fsigmoid(acc[1][tn][r]);
                float gv = ftanh(acc[2][tn][r]);
                float ov = fsigmoid(acc[3][tn][r]);
                float cn = fv * cst[tn][r] + iv * gv;
                float hv = ov * ftanh(cn);
                cst[tn][r]  = cn;
                hreg[tn][r] = hv;
                h_lds[r + half * 8][j] = f2bf(hv);
            }
        }
        __syncthreads();   // h ready for next step
    }

    // ---- out[b] = h_final . W_lin + b_lin (LDS reduction) ------------------
#pragma unroll
    for (int tn = 0; tn < 2; ++tn)
#pragma unroll
        for (int r = 0; r < 8; ++r)
            atomicAdd(&out_acc[r + half * 8], hreg[tn][r] * wlin[tn]);
    __syncthreads();
    if (tid < ROWS) out[b0 + tid] = out_acc[tid] + b_lin[0];
}

// ---- launch ----------------------------------------------------------------
extern "C" void kernel_launch(void* const* d_in, const int* in_sizes, int n_in,
                              void* d_out, int out_size, void* d_ws, size_t ws_size,
                              hipStream_t stream) {
    const float* inp   = (const float*)d_in[0];
    const float* W_ih  = (const float*)d_in[1];
    const float* W_hh  = (const float*)d_in[2];
    const float* b_ih  = (const float*)d_in[3];
    const float* b_hh  = (const float*)d_in[4];
    const float* W_lin = (const float*)d_in[5];
    const float* b_lin = (const float*)d_in[6];
    float* out = (float*)d_out;

    unsigned short* Bh   = (unsigned short*)((char*)d_ws + WS_BH_OFF);
    unsigned short* Bx   = (unsigned short*)((char*)d_ws + WS_BX_OFF);
    float*          bias = (float*)((char*)d_ws + WS_BIAS_OFF);

    // pack weights (262144 elements -> 1024 blocks x 256)
    lstm_pack_kernel<<<dim3((NG * HID + 255) / 256), dim3(256), 0, stream>>>(
        W_ih, W_hh, b_ih, b_hh, Bh, Bx, bias);

    // persistent fused scan: 64 WGs x 256 threads
    lstm_scan_kernel<<<dim3(NWG), dim3(256), 0, stream>>>(
        inp, Bh, Bx, bias, W_lin, b_lin, out);
}
